// Attention_5368709120299
// MI455X (gfx1250) — matrix operand order repaired and checked by reading, not compiled, not run on hardware
//
#include <hip/hip_runtime.h>
#include <hip/hip_bf16.h>

// ---------------------------------------------------------------------------
// CDNA5 (gfx1250) fused gated multi-head attention
//   wave32, v_wmma_f32_16x16x32_bf16, flash softmax with DPP lane reductions
// ---------------------------------------------------------------------------

typedef __bf16 bf16_t;
typedef __attribute__((ext_vector_type(16))) __bf16 v16bf;
typedef __attribute__((ext_vector_type(8)))  float  v8f;

#define NUM_HEAD 8
#define HDIM     32
#define SEQ      384   // Q == K length
#define DMODEL   256
#define NBATCH   128
#define LOG2E    1.4426950408889634f

// D = A(16x32 bf16) * B(32x16 bf16) + C(16x16 f32)
__device__ __forceinline__ v8f wmma_bf16f32(v16bf a, v16bf b, v8f c) {
  return __builtin_amdgcn_wmma_f32_16x16x32_bf16(
      /*neg_a=*/false, a, /*neg_b=*/false, b,
      /*c_mod=*/(short)0, c, /*reuse_a=*/false, /*reuse_b=*/false);
}

// ---- DPP 16-lane all-reduce (pure VALU, no LDS/bpermute) -------------------
// 0xB1 = quad_perm(1,0,3,2) (xor1), 0x4E = quad_perm(2,3,0,1) (xor2),
// 0x141 = row_half_mirror (i^7), 0x140 = row_mirror (i^15).
// DPP rows are 16 lanes -> lanes 16-31 reduce independently (matches row split).
template <int CTRL>
__device__ __forceinline__ float dpp_f32(float x) {
  return __int_as_float(__builtin_amdgcn_update_dpp(
      0, __float_as_int(x), CTRL, 0xF, 0xF, true));
}
__device__ __forceinline__ float row16_allmax(float x) {
  x = fmaxf(x, dpp_f32<0xB1>(x));
  x = fmaxf(x, dpp_f32<0x4E>(x));
  x = fmaxf(x, dpp_f32<0x141>(x));
  x = fmaxf(x, dpp_f32<0x140>(x));
  return x;
}
__device__ __forceinline__ float row16_allsum(float x) {
  x += dpp_f32<0xB1>(x);
  x += dpp_f32<0x4E>(x);
  x += dpp_f32<0x141>(x);
  x += dpp_f32<0x140>(x);
  return x;
}

// Two contiguous 16-byte chunks -> one v16bf fragment (2x b128).
__device__ __forceinline__ v16bf ld16x2(const bf16_t* p0, const bf16_t* p1) {
  union { uint4 u[2]; v16bf v; } cv;
  cv.u[0] = *reinterpret_cast<const uint4*>(p0);
  cv.u[1] = *reinterpret_cast<const uint4*>(p1);
  return cv.v;
}

// A fragment (16x32), row-major rows of `ldr` bf16.
// lanes 0-15: row=lane, K {0..7,16..23}; lanes 16-31: row=lane-16, K {8..15,24..31}
__device__ __forceinline__ v16bf load_a_frag_bf16(const bf16_t* base, int ldr) {
  const int lane = threadIdx.x & 31;
  const bf16_t* rowp = base + (size_t)(lane & 15) * ldr + ((lane >> 4) * 8);
  return ld16x2(rowp, rowp + 16);
}

// A fragment from f32 source (converted to bf16 on load).
__device__ __forceinline__ v16bf load_a_frag_f32(const float* base, int ldr) {
  const int lane = threadIdx.x & 31;
  const float* rowp = base + (size_t)(lane & 15) * ldr + ((lane >> 4) * 8);
  v16bf a;
#pragma unroll
  for (int i = 0; i < 8; ++i) a[i] = (__bf16)rowp[i];
#pragma unroll
  for (int i = 0; i < 8; ++i) a[8 + i] = (__bf16)rowp[16 + i];
  return a;
}

// B fragment (32x16) from "column-major" storage colmaj[n][k], contiguous in k.
// lanes 0-15: col=lane rows {0..7,16..23}; lanes 16-31: col=lane-16 rows {8..15,24..31}
__device__ __forceinline__ v16bf load_b_frag(const bf16_t* colmaj, int ldc) {
  const int lane = threadIdx.x & 31;
  const bf16_t* p = colmaj + (size_t)(lane & 15) * ldc + ((lane >> 4) * 8);
  return ld16x2(p, p + 16);
}

// ---------------------------------------------------------------------------
// Weight prep: w[in=256][out=256] f32 -> wT[out][in] bf16 (q gets 1/sqrt(32))
// ---------------------------------------------------------------------------
__global__ void prep_weights(const float* __restrict__ qw, const float* __restrict__ kw,
                             const float* __restrict__ vw, const float* __restrict__ ow,
                             const float* __restrict__ gw,
                             bf16_t* __restrict__ qwT, bf16_t* __restrict__ kwT,
                             bf16_t* __restrict__ vwT, bf16_t* __restrict__ owT,
                             bf16_t* __restrict__ gwT) {
  int t = blockIdx.x * blockDim.x + threadIdx.x;
  if (t >= 5 * 256 * 256) return;
  int which = t >> 16;
  int idx   = t & 0xFFFF;
  int i = idx >> 8;   // input (inner) index
  int o = idx & 255;  // output index (h*32+c flattened)
  const float* src; bf16_t* dst; float s = 1.0f;
  switch (which) {
    case 0:  src = qw; dst = qwT; s = 0.17677669529663687f; break; // 1/sqrt(32)
    case 1:  src = kw; dst = kwT; break;
    case 2:  src = vw; dst = vwT; break;
    case 3:  src = ow; dst = owT; break;                 // [h*32+d][out] same flatten
    default: src = gw; dst = gwT; break;
  }
  dst[o * 256 + i] = (__bf16)(src[i * 256 + o] * s);
}

// ---------------------------------------------------------------------------
// Projection GEMMs: x[B*S,256] f32 @ wT -> per-head tensors
//   MODE 0: q  -> [B,H,S,32] bf16 (scale in weights)   MODE 1: k -> [B,H,S,32]
//   MODE 2: v  -> [B,H,32,S] bf16 (transposed)         MODE 3: gate -> [B,S,256]
// Each wave: one 16-row x 32-col tile (A fragment reused across 2 WMMAs).
// ---------------------------------------------------------------------------
template <int MODE>
__global__ void proj_kernel(const float* __restrict__ x, const bf16_t* __restrict__ wT,
                            const float* __restrict__ gating_b, bf16_t* __restrict__ out) {
  const int lane = threadIdx.x & 31;
  const int wv   = threadIdx.x >> 5;
  const int row0 = blockIdx.x * 16;   // 16-row tile (384 % 16 == 0: no batch crossing)
  const int col0 = wv * 32;           // 32 columns per wave

  v8f acc0 = {}, acc1 = {};
#pragma unroll
  for (int k0 = 0; k0 < DMODEL; k0 += 32) {
    v16bf a  = load_a_frag_f32(x + (size_t)row0 * DMODEL + k0, DMODEL);
    v16bf b0 = load_b_frag(wT + (size_t)col0 * DMODEL + k0, DMODEL);
    v16bf b1 = load_b_frag(wT + (size_t)(col0 + 16) * DMODEL + k0, DMODEL);
    acc0 = wmma_bf16f32(a, b0, acc0);
    acc1 = wmma_bf16f32(a, b1, acc1);
  }

  const int b_idx = row0 / SEQ;
  const int pos0  = row0 % SEQ;
  const int hi8   = (lane >> 4) * 8;
  const int nc    = lane & 15;
#pragma unroll
  for (int half = 0; half < 2; ++half) {
    const int colc = col0 + half * 16 + nc;
    const int h    = colc >> 5;
    const int d    = colc & 31;
#pragma unroll
    for (int r = 0; r < 8; ++r) {
      const int pos = pos0 + r + hi8;
      const float v = half ? acc1[r] : acc0[r];
      if (MODE == 0 || MODE == 1) {
        out[(((size_t)b_idx * NUM_HEAD + h) * SEQ + pos) * HDIM + d] = (__bf16)v;
      } else if (MODE == 2) {
        out[(((size_t)b_idx * NUM_HEAD + h) * HDIM + d) * SEQ + pos] = (__bf16)v;
      } else {
        const float g = 1.0f / (1.0f + __expf(-(v + gating_b[colc])));
        out[((size_t)b_idx * SEQ + pos) * DMODEL + colc] = (__bf16)g;
      }
    }
  }
}

// ---------------------------------------------------------------------------
// Flash attention: one wave per (b, h, 16-row q tile); online softmax over K
// in the exp2 domain. Per 32-wide k step: 2 WMMA for S, bias add (*log2e),
// DPP row reductions, P through per-wave LDS (C/D -> A layout), 2 WMMA for PV.
// K fragments die after the S WMMAs; V fragments loaded just-in-time --
// keeps liveness low (round-2 double buffering caused scratch spills).
// ---------------------------------------------------------------------------
__global__ void flash_attn_kernel(const bf16_t* __restrict__ qbuf,
                                  const bf16_t* __restrict__ kbuf,
                                  const bf16_t* __restrict__ vtbuf,
                                  const bf16_t* __restrict__ gatebuf,
                                  const float* __restrict__ bias,
                                  const float* __restrict__ nbias,
                                  bf16_t* __restrict__ wabuf) {
  __shared__ __align__(16) bf16_t pstage[8][16 * 32];  // 1KB per wave
  const int lane = threadIdx.x & 31;
  const int wv   = threadIdx.x >> 5;
  const int work = blockIdx.x * 8 + wv;                // 0 .. 128*8*24-1
  const int b    = work / (NUM_HEAD * (SEQ / 16));
  const int rem  = work % (NUM_HEAD * (SEQ / 16));
  const int h    = rem / (SEQ / 16);
  const int q0   = (rem % (SEQ / 16)) * 16;

  const bf16_t* qh = qbuf  + (((size_t)b * NUM_HEAD + h) * SEQ + q0) * HDIM;
  const bf16_t* kh = kbuf  + ((size_t)b * NUM_HEAD + h) * SEQ * HDIM;
  const bf16_t* vt = vtbuf + ((size_t)b * NUM_HEAD + h) * HDIM * SEQ;

  const int hi8 = (lane >> 4) * 8;
  const int nc  = lane & 15;

  // loop-invariant bias bases (per-lane row/col added in the loop)
  const float* bias_row = bias  + (size_t)b * SEQ * SEQ + (size_t)(q0 + nc) * 0;
  const float* nb_row   = nbias + (size_t)h * SEQ * SEQ;
  const float* bias_b   = bias  + (size_t)b * SEQ * SEQ;
  // prefetch base: 32 lanes cover 16 rows x both 16-col halves of next tile
  const float* pf_b = bias_b + (size_t)(q0 + nc) * SEQ + ((lane >> 4) * 16);
  const float* pf_n = nb_row + (size_t)(q0 + nc) * SEQ + ((lane >> 4) * 16);
  (void)bias_row;

  const v16bf qA = load_a_frag_bf16(qh, HDIM);

  float m[8], l[8];
  v8f O0 = {}, O1 = {};
#pragma unroll
  for (int r = 0; r < 8; ++r) { m[r] = -3.0e38f; l[r] = 0.0f; }

  bf16_t* pb = pstage[wv];

  for (int k0 = 0; k0 < SEQ; k0 += 32) {
    const int kn = k0 + 32;
    if (kn < SEQ) {  // prefetch next step's bias lines (one inst covers 16 rows)
      __builtin_prefetch(pf_b + kn, 0, 1);
      __builtin_prefetch(pf_n + kn, 0, 1);
    }

    v8f S0, S1;
    {
      const v16bf kB0 = load_b_frag(kh + (size_t)k0 * HDIM, HDIM);
      const v16bf kB1 = load_b_frag(kh + (size_t)(k0 + 16) * HDIM, HDIM);
      v8f z = {};
      S0 = wmma_bf16f32(qA, kB0, z);
      S1 = wmma_bf16f32(qA, kB1, z);
    }

    // additive biases, scaled into exp2 domain
#pragma unroll
    for (int r = 0; r < 8; ++r) {
      const int row = q0 + r + hi8;
      const size_t o0 = (size_t)row * SEQ + k0 + nc;
      S0[r] = (S0[r] + bias_b[o0]      + nb_row[o0])      * LOG2E;
      S1[r] = (S1[r] + bias_b[o0 + 16] + nb_row[o0 + 16]) * LOG2E;
    }

    // online softmax update; each 16-lane half owns rows r (+8 for hi half)
#pragma unroll
    for (int r = 0; r < 8; ++r) {
      const float t    = row16_allmax(fmaxf(S0[r], S1[r]));
      const float mnew = fmaxf(m[r], t);
      const float alpha = exp2f(m[r] - mnew);
      const float p0 = exp2f(S0[r] - mnew);
      const float p1 = exp2f(S1[r] - mnew);
      l[r] = l[r] * alpha + row16_allsum(p0 + p1);
      m[r] = mnew;
      O0[r] *= alpha;
      O1[r] *= alpha;
      pb[(r + hi8) * 32 + nc]      = (__bf16)p0;   // re-stripe C/D -> A layout via LDS
      pb[(r + hi8) * 32 + 16 + nc] = (__bf16)p1;
    }

    // V fragments loaded just-in-time; pA after LDS writes (same-wave, in-order)
    const v16bf vB0 = load_b_frag(vt + k0, SEQ);
    const v16bf vB1 = load_b_frag(vt + (size_t)16 * SEQ + k0, SEQ);
    const v16bf pA  = load_a_frag_bf16(pb, 32);
    O0 = wmma_bf16f32(pA, vB0, O0);
    O1 = wmma_bf16f32(pA, vB1, O1);
  }

  // normalize, gate, store gated weighted_avg as bf16 [B*S, 256] (h-major cols)
#pragma unroll
  for (int r = 0; r < 8; ++r) {
    const float inv = 1.0f / l[r];
    const int row = q0 + r + hi8;
    const size_t base = ((size_t)b * SEQ + row) * DMODEL + h * HDIM;
    const float g0 = (float)gatebuf[base + nc];
    const float g1 = (float)gatebuf[base + 16 + nc];
    wabuf[base + nc]      = (__bf16)(O0[r] * inv * g0);
    wabuf[base + 16 + nc] = (__bf16)(O1[r] * inv * g1);
  }
}

// ---------------------------------------------------------------------------
// Output projection: wa[B*S,256] bf16 @ owT + o_bias -> f32 out
// ---------------------------------------------------------------------------
__global__ void out_proj_kernel(const bf16_t* __restrict__ wabuf,
                                const bf16_t* __restrict__ owT,
                                const float* __restrict__ o_bias,
                                float* __restrict__ out) {
  const int lane = threadIdx.x & 31;
  const int wv   = threadIdx.x >> 5;
  const int row0 = blockIdx.x * 16;
  const int col0 = wv * 32;

  v8f acc0 = {}, acc1 = {};
#pragma unroll
  for (int k0 = 0; k0 < DMODEL; k0 += 32) {
    v16bf a  = load_a_frag_bf16(wabuf + (size_t)row0 * DMODEL + k0, DMODEL);
    v16bf b0 = load_b_frag(owT + (size_t)col0 * DMODEL + k0, DMODEL);
    v16bf b1 = load_b_frag(owT + (size_t)(col0 + 16) * DMODEL + k0, DMODEL);
    acc0 = wmma_bf16f32(a, b0, acc0);
    acc1 = wmma_bf16f32(a, b1, acc1);
  }

  const int nc  = lane & 15;
  const int hi8 = (lane >> 4) * 8;
#pragma unroll
  for (int half = 0; half < 2; ++half) {
    const int colc = col0 + half * 16 + nc;
    const float ob = o_bias[colc];
#pragma unroll
    for (int r = 0; r < 8; ++r) {
      out[(size_t)(row0 + r + hi8) * DMODEL + colc] = (half ? acc1[r] : acc0[r]) + ob;
    }
  }
}

// ---------------------------------------------------------------------------
extern "C" void kernel_launch(void* const* d_in, const int* in_sizes, int n_in,
                              void* d_out, int out_size, void* d_ws, size_t ws_size,
                              hipStream_t stream) {
  (void)in_sizes; (void)n_in; (void)out_size; (void)ws_size;
  const float* q_data = (const float*)d_in[0];
  const float* m_data = (const float*)d_in[1];
  const float* bias   = (const float*)d_in[2];
  const float* nbias  = (const float*)d_in[3];
  const float* q_w    = (const float*)d_in[4];
  const float* k_w    = (const float*)d_in[5];
  const float* v_w    = (const float*)d_in[6];
  const float* o_w    = (const float*)d_in[7];
  const float* o_b    = (const float*)d_in[8];
  const float* g_w    = (const float*)d_in[9];
  const float* g_b    = (const float*)d_in[10];

  char* ws = (char*)d_ws;
  size_t off = 0;
  auto alloc = [&](size_t bytes) -> char* {
    char* p = ws + off;
    off = (off + bytes + 255) & ~(size_t)255;
    return p;
  };
  const size_t nBHS = (size_t)NBATCH * NUM_HEAD * SEQ * HDIM;
  const size_t nBSD = (size_t)NBATCH * SEQ * DMODEL;
  bf16_t* qbuf    = (bf16_t*)alloc(nBHS * 2);
  bf16_t* kbuf    = (bf16_t*)alloc(nBHS * 2);
  bf16_t* vtbuf   = (bf16_t*)alloc(nBHS * 2);
  bf16_t* gatebuf = (bf16_t*)alloc(nBSD * 2);
  bf16_t* wabuf   = (bf16_t*)alloc(nBSD * 2);
  bf16_t* qwT     = (bf16_t*)alloc(256 * 256 * 2);
  bf16_t* kwT     = (bf16_t*)alloc(256 * 256 * 2);
  bf16_t* vwT     = (bf16_t*)alloc(256 * 256 * 2);
  bf16_t* owT     = (bf16_t*)alloc(256 * 256 * 2);
  bf16_t* gwT     = (bf16_t*)alloc(256 * 256 * 2);

  prep_weights<<<(5 * 256 * 256 + 255) / 256, 256, 0, stream>>>(
      q_w, k_w, v_w, o_w, g_w, qwT, kwT, vwT, owT, gwT);

  const dim3 gproj((NBATCH * SEQ) / 16);  // 3072 blocks, 8 waves x 32 cols each
  proj_kernel<0><<<gproj, 256, 0, stream>>>(q_data, qwT, nullptr, qbuf);
  proj_kernel<1><<<gproj, 256, 0, stream>>>(m_data, kwT, nullptr, kbuf);
  proj_kernel<2><<<gproj, 256, 0, stream>>>(m_data, vwT, nullptr, vtbuf);
  proj_kernel<3><<<gproj, 256, 0, stream>>>(q_data, gwT, g_b, gatebuf);

  const int attn_waves = NBATCH * NUM_HEAD * (SEQ / 16);  // 24576
  flash_attn_kernel<<<attn_waves / 8, 256, 0, stream>>>(
      qbuf, kbuf, vtbuf, gatebuf, bias, nbias, wabuf);

  out_proj_kernel<<<gproj, 256, 0, stream>>>(wabuf, owT, o_b, (float*)d_out);
}
